// RNN_69106023793008
// MI455X (gfx1250) — matrix operand order repaired and checked by reading, not compiled, not run on hardware
//
#include <hip/hip_runtime.h>
#include <hip/hip_bf16.h>

// ---------------------------------------------------------------------------
// BiGRU sentence-pair classifier for MI455X (gfx1250, wave32, WMMA bf16).
//   V=32000 E=300(->320 pad) H=512 C=3 B=128 T=128
// Pipeline:
//   1. convert weights f32->bf16 (Wi padded E->320), gather embeddings->bf16
//   2. 128 graph-captured step launches; each step: 4 independent scans
//      (fw/bw x seq p/h), fused GEMM  gates = h_prev@Wh^T + x_t@Wi^T via
//      v_wmma_f32_16x16x32_bf16, then GRU gate math, ping-pong h buffers.
//   3. fc1 (WMMA, M=128 N=512 K=2048) + ReLU, fc2 (C=3, plain VALU).
// ---------------------------------------------------------------------------

typedef __bf16 bf16_t;
typedef bf16_t v16bf __attribute__((ext_vector_type(16)));
typedef float  v8f   __attribute__((ext_vector_type(8)));

union V16U { v16bf v; uint4 q[2]; };

#define B_  128
#define T_  128
#define ESR 300
#define EP  320
#define H_  512
#define G_  1536   // 3H
#define ZD  2048   // 4H (also per-row span of the 4-scan hidden buffer)

static __device__ __forceinline__ unsigned short f2bf(float f) {
    union { float f; unsigned u; } v; v.f = f;
    unsigned r = v.u + 0x7FFFu + ((v.u >> 16) & 1u);  // round-to-nearest-even
    return (unsigned short)(r >> 16);
}

static __device__ __forceinline__ v8f wmma_bf16(const V16U& a, const V16U& b, v8f c) {
    return __builtin_amdgcn_wmma_f32_16x16x32_bf16(
        /*neg_a=*/false, a.v, /*neg_b=*/false, b.v,
        /*c_mod=*/(short)0, c, /*reuse_a=*/false, /*reuse_b=*/false);
}

static __device__ __forceinline__ void ldA(V16U& a, const unsigned short* p) {
    // lane-local A chunks: [p+0..7] and [p+8..15]  (16B each)
    a.q[0] = *reinterpret_cast<const uint4*>(p);
    a.q[1] = *reinterpret_cast<const uint4*>(p + 8);
}
static __device__ __forceinline__ void ldB(V16U& b, const unsigned short* p) {
    // lane-local B chunk: 16 contiguous bf16 (32B)
    const uint4* q = reinterpret_cast<const uint4*>(p);
    b.q[0] = q[0]; b.q[1] = q[1];
}

// ---------------- conversion / gather kernels ------------------------------

__global__ void cvt_pad_bf16(const float* __restrict__ src,
                             unsigned short* __restrict__ dst,
                             int rows, int scols, int dcols) {
    int tid = blockIdx.x * blockDim.x + threadIdx.x;
    int total = rows * dcols;
    if (tid >= total) return;
    int r = tid / dcols, c = tid - r * dcols;
    float v = (c < scols) ? src[(size_t)r * scols + c] : 0.0f;
    dst[tid] = f2bf(v);
}

__global__ void gather_embed_bf16(const int* __restrict__ idx,
                                  const float* __restrict__ emb,
                                  unsigned short* __restrict__ dst, int n) {
    int tid = blockIdx.x * blockDim.x + threadIdx.x;
    if (tid >= n) return;
    int e  = tid % EP;
    int bt = tid / EP;            // b*T + t
    float v = (e < ESR) ? emb[(size_t)idx[bt] * ESR + e] : 0.0f;
    dst[tid] = f2bf(v);
}

// ---------------- fused GRU step (one timestep, all 4 scans) ---------------
// grid = (32 column-strips, 4 scans), block = 256 (8 waves; wave w -> M-tile w)

__global__ void __launch_bounds__(256, 1)
gru_step(const unsigned short* __restrict__ Xb,    // [2][B][T][EP] bf16
         const unsigned short* __restrict__ Wib,   // [2][G][EP]    bf16
         const unsigned short* __restrict__ Whb,   // [2][G][H]     bf16
         const float* __restrict__ fw_bi, const float* __restrict__ fw_bh,
         const float* __restrict__ bw_bi, const float* __restrict__ bw_bh,
         const float* __restrict__ hprev_f32,      // [B][ZD]
         const unsigned short* __restrict__ hprev_b16,
         float* __restrict__ hnext_f32,
         unsigned short* __restrict__ hnext_b16,
         int t) {
    const int strip = blockIdx.x;            // 0..31 -> 16 columns of H
    const int s     = blockIdx.y;            // scan: 0 fw_x, 1 bw_x, 2 fw_y, 3 bw_y
    const int wave  = threadIdx.x >> 5;      // M-tile
    const int lane  = threadIdx.x & 31;
    const int lanelo = lane & 15;
    const int hi     = lane >> 4;            // 0: lanes 0-15, 1: lanes 16-31
    const int kbaseA = hi ? 16 : 0;
    const int k0B    = hi ? 16 : 0;

    const int seq  = s >> 1;
    const int dir  = s & 1;
    const int teff = dir ? (T_ - 1 - t) : t;
    const int colg = strip * 16 + lanelo;    // column in H, also B-matrix N row

    const unsigned short* Arow_h = hprev_b16 + (size_t)(wave * 16 + lanelo) * ZD + s * H_;
    const unsigned short* Arow_x = Xb +
        (((size_t)seq * B_ + (wave * 16 + lanelo)) * T_ + teff) * EP;

    const unsigned short* Whd = Whb + (size_t)dir * G_ * H_;
    const unsigned short* Wid = Wib + (size_t)dir * G_ * EP;
    const unsigned short* Bh_r = Whd + (size_t)(0 * H_ + colg) * H_ + k0B;
    const unsigned short* Bh_z = Whd + (size_t)(1 * H_ + colg) * H_ + k0B;
    const unsigned short* Bh_n = Whd + (size_t)(2 * H_ + colg) * H_ + k0B;
    const unsigned short* Bx_r = Wid + (size_t)(0 * H_ + colg) * EP + k0B;
    const unsigned short* Bx_z = Wid + (size_t)(1 * H_ + colg) * EP + k0B;
    const unsigned short* Bx_n = Wid + (size_t)(2 * H_ + colg) * EP + k0B;

    v8f accR  = {0, 0, 0, 0, 0, 0, 0, 0};
    v8f accZ  = {0, 0, 0, 0, 0, 0, 0, 0};
    v8f accIN = {0, 0, 0, 0, 0, 0, 0, 0};
    v8f accHN = {0, 0, 0, 0, 0, 0, 0, 0};

    // hidden-to-hidden: K = 512
    for (int kk = 0; kk < H_; kk += 32) {
        V16U a, br, bz, bn;
        ldA(a, Arow_h + kk + kbaseA);
        ldB(br, Bh_r + kk); ldB(bz, Bh_z + kk); ldB(bn, Bh_n + kk);
        accR  = wmma_bf16(a, br, accR);
        accZ  = wmma_bf16(a, bz, accZ);
        accHN = wmma_bf16(a, bn, accHN);
    }
    // input-to-hidden: K = 320 (E padded with zeros)
    for (int kk = 0; kk < EP; kk += 32) {
        V16U a, br, bz, bn;
        ldA(a, Arow_x + kk + kbaseA);
        ldB(br, Bx_r + kk); ldB(bz, Bx_z + kk); ldB(bn, Bx_n + kk);
        accR  = wmma_bf16(a, br, accR);
        accZ  = wmma_bf16(a, bz, accZ);
        accIN = wmma_bf16(a, bn, accIN);
    }

    const float* bi = dir ? bw_bi : fw_bi;
    const float* bh = dir ? bw_bh : fw_bh;
    const float brr = bi[colg]          + bh[colg];
    const float bzz = bi[H_ + colg]     + bh[H_ + colg];
    const float bin = bi[2 * H_ + colg];
    const float bhn = bh[2 * H_ + colg];

    #pragma unroll
    for (int e = 0; e < 8; ++e) {
        const int mg = wave * 16 + (hi ? (8 + e) : e);
        const size_t oidx = (size_t)mg * ZD + s * H_ + colg;
        const float hp = hprev_f32[oidx];
        const float r  = 1.0f / (1.0f + __expf(-(accR[e] + brr)));
        const float z  = 1.0f / (1.0f + __expf(-(accZ[e] + bzz)));
        const float n  = tanhf((accIN[e] + bin) + r * (accHN[e] + bhn));
        const float hn = (1.0f - z) * n + z * hp;
        hnext_f32[oidx] = hn;
        hnext_b16[oidx] = f2bf(hn);
    }
}

// ---------------- fc1: [128,2048] x [2048,512]^T + bias, ReLU --------------
// grid = 32 strips, block = 256 (8 waves -> 8 M-tiles)

__global__ void __launch_bounds__(256, 1)
fc1_wmma(const unsigned short* __restrict__ zb16,   // [B][ZD] bf16 (final h)
         const unsigned short* __restrict__ fc1b,   // [H][ZD] bf16
         const float* __restrict__ fc1_bias,
         float* __restrict__ out) {                 // [B][H] f32
    const int strip = blockIdx.x;
    const int wave  = threadIdx.x >> 5;
    const int lane  = threadIdx.x & 31;
    const int lanelo = lane & 15;
    const int hi     = lane >> 4;
    const int kbaseA = hi ? 16 : 0;
    const int k0B    = hi ? 16 : 0;
    const int ng = strip * 16 + lanelo;

    const unsigned short* Arow = zb16 + (size_t)(wave * 16 + lanelo) * ZD;
    const unsigned short* Brow = fc1b + (size_t)ng * ZD + k0B;

    v8f acc = {0, 0, 0, 0, 0, 0, 0, 0};
    for (int kk = 0; kk < ZD; kk += 32) {
        V16U a, b;
        ldA(a, Arow + kk + kbaseA);
        ldB(b, Brow + kk);
        acc = wmma_bf16(a, b, acc);
    }
    const float bias = fc1_bias[ng];
    #pragma unroll
    for (int e = 0; e < 8; ++e) {
        const int mg = wave * 16 + (hi ? (8 + e) : e);
        float v = acc[e] + bias;
        out[(size_t)mg * H_ + ng] = v > 0.0f ? v : 0.0f;
    }
}

// ---------------- fc2: [128,512] x [512,3]^T + bias ------------------------

__global__ void fc2_small(const float* __restrict__ x,      // [B][H]
                          const float* __restrict__ w,      // [3][H]
                          const float* __restrict__ b,
                          float* __restrict__ out) {        // [B][3]
    const int bi = threadIdx.x;            // 0..127
    if (bi >= B_) return;
    #pragma unroll
    for (int c = 0; c < 3; ++c) {
        float acc = b[c];
        const float* xr = x + (size_t)bi * H_;
        const float* wr = w + (size_t)c * H_;
        for (int k = 0; k < H_; ++k) acc += xr[k] * wr[k];
        out[bi * 3 + c] = acc;
    }
}

// ---------------------------------------------------------------------------

extern "C" void kernel_launch(void* const* d_in, const int* in_sizes, int n_in,
                              void* d_out, int out_size, void* d_ws, size_t ws_size,
                              hipStream_t stream) {
    const int*   p      = (const int*)  d_in[0];
    const int*   hq     = (const int*)  d_in[1];
    const float* emb    = (const float*)d_in[2];
    const float* fw_Wi  = (const float*)d_in[3];
    const float* fw_Wh  = (const float*)d_in[4];
    const float* fw_bi  = (const float*)d_in[5];
    const float* fw_bh  = (const float*)d_in[6];
    const float* bw_Wi  = (const float*)d_in[7];
    const float* bw_Wh  = (const float*)d_in[8];
    const float* bw_bi  = (const float*)d_in[9];
    const float* bw_bh  = (const float*)d_in[10];
    const float* fc1_w  = (const float*)d_in[11];
    const float* fc1_b  = (const float*)d_in[12];
    const float* fc2_w  = (const float*)d_in[13];
    const float* fc2_b  = (const float*)d_in[14];
    float* out = (float*)d_out;

    // ---- workspace carve-out (256B aligned) ----
    char* ws = (char*)d_ws;
    size_t off = 0;
    auto carve = [&](size_t bytes) -> char* {
        char* q = ws + off;
        off = (off + bytes + 255) & ~(size_t)255;
        return q;
    };
    unsigned short* Xb   = (unsigned short*)carve((size_t)2 * B_ * T_ * EP * 2);  // 21 MB
    unsigned short* Wib  = (unsigned short*)carve((size_t)2 * G_ * EP * 2);
    unsigned short* Whb  = (unsigned short*)carve((size_t)2 * G_ * H_ * 2);
    unsigned short* fc1b = (unsigned short*)carve((size_t)H_ * ZD * 2);
    float*          hf0  = (float*)         carve((size_t)B_ * ZD * 4);
    float*          hf1  = (float*)         carve((size_t)B_ * ZD * 4);
    unsigned short* hb0  = (unsigned short*)carve((size_t)B_ * ZD * 2);
    unsigned short* hb1  = (unsigned short*)carve((size_t)B_ * ZD * 2);
    float*          fco  = (float*)         carve((size_t)B_ * H_ * 4);

    const int TPB = 256;
    auto blocks = [](size_t n, int tpb) { return (unsigned)((n + tpb - 1) / tpb); };

    // ---- weight conversion ----
    cvt_pad_bf16<<<blocks((size_t)G_ * EP, TPB), TPB, 0, stream>>>(fw_Wi, Wib,               G_, ESR, EP);
    cvt_pad_bf16<<<blocks((size_t)G_ * EP, TPB), TPB, 0, stream>>>(bw_Wi, Wib + (size_t)G_ * EP, G_, ESR, EP);
    cvt_pad_bf16<<<blocks((size_t)G_ * H_, TPB), TPB, 0, stream>>>(fw_Wh, Whb,               G_, H_, H_);
    cvt_pad_bf16<<<blocks((size_t)G_ * H_, TPB), TPB, 0, stream>>>(bw_Wh, Whb + (size_t)G_ * H_, G_, H_, H_);
    cvt_pad_bf16<<<blocks((size_t)H_ * ZD, TPB), TPB, 0, stream>>>(fc1_w, fc1b,              H_, ZD, ZD);

    // ---- embedding gather (both sequences) ----
    const size_t xn = (size_t)B_ * T_ * EP;
    gather_embed_bf16<<<blocks(xn, TPB), TPB, 0, stream>>>(p,  emb, Xb,      (int)xn);
    gather_embed_bf16<<<blocks(xn, TPB), TPB, 0, stream>>>(hq, emb, Xb + xn, (int)xn);

    // ---- h0 = 0 ----
    hipMemsetAsync(hf0, 0, (size_t)B_ * ZD * 4, stream);
    hipMemsetAsync(hb0, 0, (size_t)B_ * ZD * 2, stream);

    // ---- 128 recurrent steps (4 independent scans per step) ----
    float*          hf[2] = {hf0, hf1};
    unsigned short* hb[2] = {hb0, hb1};
    int cur = 0;
    dim3 sgrid(H_ / 16, 4);
    for (int t = 0; t < T_; ++t) {
        gru_step<<<sgrid, 256, 0, stream>>>(Xb, Wib, Whb,
                                            fw_bi, fw_bh, bw_bi, bw_bh,
                                            hf[cur], hb[cur],
                                            hf[1 - cur], hb[1 - cur], t);
        cur ^= 1;
    }

    // ---- classifier ----
    fc1_wmma<<<H_ / 16, 256, 0, stream>>>(hb[cur], fc1b, fc1_b, fco);
    fc2_small<<<1, 128, 0, stream>>>(fco, fc2_w, fc2_b, out);
}
